// AyanamiT2DecoderLayer_17952963297413
// MI455X (gfx1250) — compile-verified, matrix operand order checked by reading, not run on hardware
//
#include <hip/hip_runtime.h>
#include <hip/hip_bf16.h>

// ---------------- problem constants ----------------
#define BB   2
#define SS   2048
#define DD   1024
#define HH   16
#define DHD  64
#define DFF2 4096
#define NEGV (-1e9f)

typedef __bf16 bf16;
typedef __attribute__((ext_vector_type(16))) __bf16 bf16x16;
typedef __attribute__((ext_vector_type(8)))  float  f32x8;

#define WMMA_BF16(a, b, c) \
  __builtin_amdgcn_wmma_f32_16x16x32_bf16(false, (a), false, (b), (short)0, (c), false, false)

// ---------------- CDNA5 async copy (global -> LDS, ASYNCcnt-tracked) ----------------
// LDS destination comes from the vdst VGPR; low 32 bits of a flat shared-mem address are the
// wave-relative LDS byte offset (ISA 10.2).
__device__ __forceinline__ uint32_t lds_addr32(const void* p) {
  return (uint32_t)(uintptr_t)p;
}
__device__ __forceinline__ void async_copy_b128(uint32_t lds_byte_off, const void* gptr) {
  asm volatile("global_load_async_to_lds_b128 %0, %1, off"
               :: "v"(lds_byte_off), "v"(gptr) : "memory");
}
__device__ __forceinline__ void wait_async0() {
  asm volatile("s_wait_asynccnt 0x0" ::: "memory");
}

// ---------------- small helpers ----------------
__device__ __forceinline__ float red16_max(float v) {
  v = fmaxf(v, __shfl_xor(v, 1, 32));
  v = fmaxf(v, __shfl_xor(v, 2, 32));
  v = fmaxf(v, __shfl_xor(v, 4, 32));
  v = fmaxf(v, __shfl_xor(v, 8, 32));
  return v;
}
__device__ __forceinline__ float red16_sum(float v) {
  v += __shfl_xor(v, 1, 32);
  v += __shfl_xor(v, 2, 32);
  v += __shfl_xor(v, 4, 32);
  v += __shfl_xor(v, 8, 32);
  return v;
}
__device__ __forceinline__ float gelu_exact(float x) {
  return 0.5f * x * (1.0f + erff(x * 0.70710678118654752f));
}

// ---------------- fp32 -> bf16 cast ----------------
__global__ __launch_bounds__(256) void cast_bf16_kernel(const float* __restrict__ src,
                                                        bf16* __restrict__ dst, int n) {
  int i = blockIdx.x * 256 + threadIdx.x;
  int stride = gridDim.x * 256;
  for (; i < n; i += stride) dst[i] = (bf16)src[i];
}

// ---------------- LayerNorm (fp32 in, bf16 out), D = 1024, 256 thr/row ----------------
__global__ __launch_bounds__(256) void ln_kernel(const float* __restrict__ X,
                                                 const float* __restrict__ g,
                                                 const float* __restrict__ b,
                                                 bf16* __restrict__ out) {
  __shared__ float sa[256], sb[256];
  int row = blockIdx.x;
  int t = threadIdx.x;
  const float* xr = X + (size_t)row * DD;
  float4 xv = *(const float4*)(xr + t * 4);
  float ls = xv.x + xv.y + xv.z + xv.w;
  float lq = xv.x * xv.x + xv.y * xv.y + xv.z * xv.z + xv.w * xv.w;
  sa[t] = ls; sb[t] = lq;
  __syncthreads();
  for (int off = 128; off > 0; off >>= 1) {
    if (t < off) { sa[t] += sa[t + off]; sb[t] += sb[t + off]; }
    __syncthreads();
  }
  float mu = sa[0] * (1.0f / DD);
  float var = sb[0] * (1.0f / DD) - mu * mu;
  float rstd = rsqrtf(var + 1e-5f);
  float4 gv = *(const float4*)(g + t * 4);
  float4 bv = *(const float4*)(b + t * 4);
  bf16* o = out + (size_t)row * DD + t * 4;
  o[0] = (bf16)((xv.x - mu) * rstd * gv.x + bv.x);
  o[1] = (bf16)((xv.y - mu) * rstd * gv.y + bv.y);
  o[2] = (bf16)((xv.z - mu) * rstd * gv.z + bv.z);
  o[3] = (bf16)((xv.w - mu) * rstd * gv.w + bv.w);
}

// ---------------- generic WMMA GEMM: C[M,N] = A[M,K] * W[N,K]^T (+bias)(+gelu)(+res) ----------------
// block: 256 thr (8 waves, 2x4), block tile 128x256, wave tile 64x64 (4x4 frags), K step 32.
// Tiles staged into double-buffered LDS via global_load_async_to_lds_b128 (ASYNCcnt).
#define LDS_STRIDE 48
__global__ __launch_bounds__(256) void gemm_bf16_kernel(const bf16* __restrict__ A,
                                                        const bf16* __restrict__ W,
                                                        const float* __restrict__ bias,
                                                        const float* __restrict__ res,
                                                        bf16* __restrict__ outb,
                                                        float* __restrict__ outf,
                                                        int M, int N, int K, int act_gelu) {
  __shared__ __attribute__((aligned(32))) bf16 As[2][128 * LDS_STRIDE];
  __shared__ __attribute__((aligned(32))) bf16 Bs[2][256 * LDS_STRIDE];

  const int t = threadIdx.x;
  const int lane = t & 31;
  const int wid = t >> 5;
  const int m0 = blockIdx.y * 128;
  const int n0 = blockIdx.x * 256;
  const int mw = (wid >> 2) * 64;   // 0 or 64
  const int nw = (wid & 3) * 64;    // 0..192
  const int fr = lane & 15;         // fragment row/col within 16
  const int fk = (lane >> 4) * 16;  // K half select
  const int arow = t >> 1;
  const int acol = (t & 1) * 16;

  // per-thread LDS destinations (byte offsets) and global sources
  const uint32_t la[2] = { lds_addr32(&As[0][arow * LDS_STRIDE + acol]),
                           lds_addr32(&As[1][arow * LDS_STRIDE + acol]) };
  const uint32_t lb[2] = { lds_addr32(&Bs[0][t * LDS_STRIDE]),
                           lds_addr32(&Bs[1][t * LDS_STRIDE]) };
  const bf16* Agbase = A + (size_t)(m0 + arow) * K + acol;
  const bf16* Wgbase = W + (size_t)(n0 + t) * K;

  f32x8 acc[4][4] = {};

  const int ksteps = K >> 5;
  // prologue: stage tile 0 into buffer 0
  {
    async_copy_b128(la[0],      Agbase);
    async_copy_b128(la[0] + 16, Agbase + 8);
    async_copy_b128(lb[0],      Wgbase);
    async_copy_b128(lb[0] + 16, Wgbase + 8);
    async_copy_b128(lb[0] + 32, Wgbase + 16);
    async_copy_b128(lb[0] + 48, Wgbase + 24);
    wait_async0();
    __syncthreads();
  }

  for (int kt = 0; kt < ksteps; ++kt) {
    const int cur = kt & 1, nxt = cur ^ 1;
    // issue async staging of next K tile into the other buffer
    if (kt + 1 < ksteps) {
      const bf16* Ag = Agbase + (size_t)(kt + 1) * 32;
      const bf16* Wg = Wgbase + (size_t)(kt + 1) * 32;
      async_copy_b128(la[nxt],      Ag);
      async_copy_b128(la[nxt] + 16, Ag + 8);
      async_copy_b128(lb[nxt],      Wg);
      async_copy_b128(lb[nxt] + 16, Wg + 8);
      async_copy_b128(lb[nxt] + 32, Wg + 16);
      async_copy_b128(lb[nxt] + 48, Wg + 24);
    }

    // compute on current buffer
    bf16x16 af[4];
#pragma unroll
    for (int i = 0; i < 4; ++i)
      af[i] = *(const bf16x16*)(&As[cur][(mw + i * 16 + fr) * LDS_STRIDE + fk]);
#pragma unroll
    for (int j = 0; j < 4; ++j) {
      bf16x16 bfj = *(const bf16x16*)(&Bs[cur][(nw + j * 16 + fr) * LDS_STRIDE + fk]);
#pragma unroll
      for (int i = 0; i < 4; ++i) acc[i][j] = WMMA_BF16(af[i], bfj, acc[i][j]);
    }

    // next tile arrived + everyone done reading current buffer
    wait_async0();
    __syncthreads();
  }

  // epilogue: C layout -> lanes 0-15: N=lane, M=r; lanes 16-31: N=lane-16, M=8+r
  const int half = lane >> 4;
  const int cl = lane & 15;
#pragma unroll
  for (int i = 0; i < 4; ++i) {
#pragma unroll
    for (int j = 0; j < 4; ++j) {
#pragma unroll
      for (int r = 0; r < 8; ++r) {
        int row = m0 + mw + i * 16 + r + 8 * half;
        int col = n0 + nw + j * 16 + cl;
        float v = acc[i][j][r];
        if (bias) v += bias[col];
        if (act_gelu) v = gelu_exact(v);
        if (res) v += res[(size_t)row * N + col];
        if (outf) outf[(size_t)row * N + col] = v;
        if (outb) outb[(size_t)row * N + col] = (bf16)v;
      }
    }
  }
}

// ---------------- V transpose: [B,S,H,DH](bf16) -> VT [B,H,DH,S](bf16) ----------------
__global__ void transpose_v_kernel(const bf16* __restrict__ V, bf16* __restrict__ VT) {
  __shared__ bf16 tile[32][33];
  int bh = blockIdx.z;
  int b = bh / HH, h = bh % HH;
  int s0 = blockIdx.x * 32, d0 = blockIdx.y * 32;
  int tx = threadIdx.x, ty = threadIdx.y;  // (32, 8)
  for (int i = ty; i < 32; i += 8)
    tile[i][tx] = V[(size_t)(b * SS + s0 + i) * DD + h * DHD + d0 + tx];
  __syncthreads();
  for (int i = ty; i < 32; i += 8)
    VT[(size_t)((b * HH + h) * DHD + d0 + i) * SS + s0 + tx] = tile[tx][i];
}

// ---------------- flash attention: 1 wave = 16 query rows ----------------
// grid (S/128, H, B), block 256 (8 waves)
__global__ __launch_bounds__(256) void attn_kernel(const bf16* __restrict__ Q,
                                                   const bf16* __restrict__ Km,
                                                   const bf16* __restrict__ VT,
                                                   const int* __restrict__ vlens,
                                                   bf16* __restrict__ O) {
  __shared__ __attribute__((aligned(32))) bf16 Plds[8][16][32];

  const int lane = threadIdx.x & 31;
  const int wid = threadIdx.x >> 5;
  const int half = lane >> 4;
  const int cl = lane & 15;
  const int b = blockIdx.z, h = blockIdx.y;
  const int qbase = blockIdx.x * 128 + wid * 16;
  const int vlen = vlens[b];
  const float rscale = 0.125f;  // 1/sqrt(64)

  // Q A-fragments: lane holds row M=cl, K chunk selected by lane half
  const bf16* qp = Q + (size_t)(b * SS + qbase + cl) * DD + h * DHD + 16 * half;
  bf16x16 qa0 = *(const bf16x16*)qp;         // dh 0..31
  bf16x16 qa1 = *(const bf16x16*)(qp + 32);  // dh 32..63

  f32x8 o[4] = {};
  float rmax[8], rsum[8];
#pragma unroll
  for (int r = 0; r < 8; ++r) { rmax[r] = -3.0e38f; rsum[r] = 0.0f; }

  const int ktiles = (qbase + 15) / 32 + 1;  // causal bound for this wave
  for (int kt = 0; kt < ktiles; ++kt) {
    const int kb = kt * 32;
    // scores for 32 keys: two 16x16 C-frags, contraction over dh (two K=32 steps)
    f32x8 s[2];
#pragma unroll
    for (int tfr = 0; tfr < 2; ++tfr) {
      const bf16* kp = Km + (size_t)(b * SS + kb + 16 * tfr + cl) * DD + h * DHD + 16 * half;
      bf16x16 kb0 = *(const bf16x16*)kp;
      bf16x16 kb1 = *(const bf16x16*)(kp + 32);
      f32x8 sc = {};
      sc = WMMA_BF16(qa0, kb0, sc);
      sc = WMMA_BF16(qa1, kb1, sc);
      s[tfr] = sc;
    }
    // masks + online softmax
#pragma unroll
    for (int r = 0; r < 8; ++r) {
      const int qrow = qbase + r + 8 * half;
      const int qpad = (qrow >= vlen);
#pragma unroll
      for (int tfr = 0; tfr < 2; ++tfr) {
        int key = kb + 16 * tfr + cl;
        float v = s[tfr][r] * rscale;
        if (key > qrow) v += NEGV;                 // causal
        if (qpad || key >= vlen) v += NEGV;        // valid-length
        s[tfr][r] = v;
      }
      float mv = red16_max(fmaxf(s[0][r], s[1][r]));
      float mnew = fmaxf(rmax[r], mv);
      float alpha = __expf(rmax[r] - mnew);
      rmax[r] = mnew;
      float p0 = __expf(s[0][r] - mnew);
      float p1 = __expf(s[1][r] - mnew);
      Plds[wid][r + 8 * half][cl]      = (bf16)p0;
      Plds[wid][r + 8 * half][16 + cl] = (bf16)p1;
      rsum[r] = rsum[r] * alpha + red16_sum(p0 + p1);
#pragma unroll
      for (int n = 0; n < 4; ++n) o[n][r] *= alpha;
    }
    // re-fetch P as an A-fragment (16x32, K=keys): lane holds row M=cl, K chunk by half
    bf16x16 pa = *(const bf16x16*)(&Plds[wid][cl][16 * half]);
#pragma unroll
    for (int n = 0; n < 4; ++n) {
      const bf16* vp = VT + (size_t)((b * HH + h) * DHD + n * 16 + cl) * SS + kb + 16 * half;
      bf16x16 vb = *(const bf16x16*)vp;
      o[n] = WMMA_BF16(pa, vb, o[n]);
    }
  }

  // finalize: divide by row sums, write bf16 [B,S,H,DH]
#pragma unroll
  for (int n = 0; n < 4; ++n) {
#pragma unroll
    for (int r = 0; r < 8; ++r) {
      int qrow = qbase + r + 8 * half;
      float v = o[n][r] / rsum[r];
      O[(size_t)(b * SS + qrow) * DD + h * DHD + n * 16 + cl] = (bf16)v;
    }
  }
}

// ---------------- host-side orchestration ----------------
extern "C" void kernel_launch(void* const* d_in, const int* in_sizes, int n_in,
                              void* d_out, int out_size, void* d_ws, size_t ws_size,
                              hipStream_t stream) {
  (void)in_sizes; (void)n_in; (void)out_size; (void)ws_size;
  const float* X     = (const float*)d_in[0];
  const int*   vlens = (const int*)d_in[2];
  const float* wq_w  = (const float*)d_in[3];
  const float* wq_b  = (const float*)d_in[4];
  const float* wk_w  = (const float*)d_in[5];
  const float* wv_w  = (const float*)d_in[6];
  const float* wv_b  = (const float*)d_in[7];
  const float* wo_w  = (const float*)d_in[8];
  const float* wo_b  = (const float*)d_in[9];
  const float* ln1_g = (const float*)d_in[10];
  const float* ln1_b = (const float*)d_in[11];
  const float* ln2_g = (const float*)d_in[12];
  const float* ln2_b = (const float*)d_in[13];
  const float* w1    = (const float*)d_in[14];
  const float* b1    = (const float*)d_in[15];
  const float* w2    = (const float*)d_in[16];
  const float* b2    = (const float*)d_in[17];
  float* out = (float*)d_out;

  // workspace carve-up (bytes)
  char* ws = (char*)d_ws;
  const size_t SZ_DD  = (size_t)DD * DD * sizeof(bf16);        // 2 MB
  const size_t SZ_DF  = (size_t)DD * DFF2 * sizeof(bf16);      // 8 MB
  const size_t SZ_ACT = (size_t)BB * SS * DD * sizeof(bf16);   // 8 MB
  bf16* wqb = (bf16*)ws;                 ws += SZ_DD;
  bf16* wkb = (bf16*)ws;                 ws += SZ_DD;
  bf16* wvb = (bf16*)ws;                 ws += SZ_DD;
  bf16* wob = (bf16*)ws;                 ws += SZ_DD;
  bf16* w1b = (bf16*)ws;                 ws += SZ_DF;
  bf16* w2b = (bf16*)ws;                 ws += SZ_DF;
  bf16* Xnb = (bf16*)ws;                 ws += SZ_ACT;  // reused as LN2 output
  bf16* Qb  = (bf16*)ws;                 ws += SZ_ACT;  // Qb..VTb reused as FFN hidden
  bf16* Kb  = (bf16*)ws;                 ws += SZ_ACT;
  bf16* Vb  = (bf16*)ws;                 ws += SZ_ACT;
  bf16* VTb = (bf16*)ws;                 ws += SZ_ACT;
  bf16* attnb = (bf16*)ws;               ws += SZ_ACT;
  float* X1 = (float*)ws;                ws += (size_t)BB * SS * DD * sizeof(float);
  bf16* Hb  = Qb;   // 32 MB region Qb..VTb reused for FFN hidden [B*S, DFF]
  bf16* Yb  = Xnb;  // LN2 output reuses LN1 buffer

  const int rows = BB * SS;           // 4096
  dim3 blk256(256);

  // 1) weights -> bf16
  cast_bf16_kernel<<<2048, blk256, 0, stream>>>(wq_w, wqb, DD * DD);
  cast_bf16_kernel<<<2048, blk256, 0, stream>>>(wk_w, wkb, DD * DD);
  cast_bf16_kernel<<<2048, blk256, 0, stream>>>(wv_w, wvb, DD * DD);
  cast_bf16_kernel<<<2048, blk256, 0, stream>>>(wo_w, wob, DD * DD);
  cast_bf16_kernel<<<4096, blk256, 0, stream>>>(w1, w1b, DD * DFF2);
  cast_bf16_kernel<<<4096, blk256, 0, stream>>>(w2, w2b, DD * DFF2);

  // 2) LN1
  ln_kernel<<<rows, blk256, 0, stream>>>(X, ln1_g, ln1_b, Xnb);

  // 3) QKV projections
  dim3 gD(DD / 256, rows / 128);  // (4, 32)
  gemm_bf16_kernel<<<gD, blk256, 0, stream>>>(Xnb, wqb, wq_b, nullptr, Qb, nullptr, rows, DD, DD, 0);
  gemm_bf16_kernel<<<gD, blk256, 0, stream>>>(Xnb, wkb, nullptr, nullptr, Kb, nullptr, rows, DD, DD, 0);
  gemm_bf16_kernel<<<gD, blk256, 0, stream>>>(Xnb, wvb, wv_b, nullptr, Vb, nullptr, rows, DD, DD, 0);

  // 4) V transpose for contiguous P*V fragments
  transpose_v_kernel<<<dim3(SS / 32, DHD / 32, BB * HH), dim3(32, 8), 0, stream>>>(Vb, VTb);

  // 5) attention
  attn_kernel<<<dim3(SS / 128, HH, BB), blk256, 0, stream>>>(Qb, Kb, VTb, vlens, attnb);

  // 6) WO projection + residual -> X1 (fp32)
  gemm_bf16_kernel<<<gD, blk256, 0, stream>>>(attnb, wob, wo_b, X, nullptr, X1, rows, DD, DD, 0);

  // 7) LN2
  ln_kernel<<<rows, blk256, 0, stream>>>(X1, ln2_g, ln2_b, Yb);

  // 8) FFN1 + exact GELU -> Hb (bf16)
  dim3 gF1(DFF2 / 256, rows / 128);  // (16, 32)
  gemm_bf16_kernel<<<gF1, blk256, 0, stream>>>(Yb, w1b, b1, nullptr, Hb, nullptr, rows, DFF2, DD, 1);

  // 9) FFN2 + residual -> out (fp32)
  gemm_bf16_kernel<<<gD, blk256, 0, stream>>>(Hb, w2b, b2, X1, nullptr, out, rows, DD, DFF2, 0);
}
